// SSIM_16097537425977
// MI455X (gfx1250) — matrix operand order repaired and checked by reading, not compiled, not run on hardware
//
#include <hip/hip_runtime.h>

// SSIM (11x11 Gaussian, sigma=1.5) on 32x1x512x512 f32 images, MI455X / gfx1250.
// Separable conv done as two banded-GEMM passes on v_wmma_f32_16x16x32_f16
// (f16 data, f32 accumulate). Per-wave 16x16 output tile; LDS transpose
// between passes; deterministic two-stage mean reduction.

typedef __attribute__((ext_vector_type(16))) _Float16 v16h;
typedef __attribute__((ext_vector_type(8)))  float    v8f;

#define IMG_H   512
#define IMG_W   512
#define BATCH   32
#define TILES   32          // 512/16 per dim
#define WPB     8           // waves per block (256 threads)
#define LSTRIDE 36          // LDS row stride (floats): conflict-free, >= 32
#define NBLOCKS ((BATCH * TILES * TILES) / WPB)   // 4096

__global__ __launch_bounds__(256)
void ssim_wmma_kernel(const float* __restrict__ img1,
                      const float* __restrict__ img2,
                      const float* __restrict__ win,   // [1,1,11,11] = outer(g,g)
                      float* __restrict__ blockSums)
{
    __shared__ float vlds[WPB][16 * LSTRIDE];
    __shared__ float redsh[WPB];

    const int tid  = threadIdx.x;
    const int wave = tid >> 5;
    const int lane = tid & 31;
    const int h    = lane >> 4;   // lane half
    const int m    = lane & 15;   // M (A/C/D) or N (B) index

    const int tileId = blockIdx.x * WPB + wave;      // 0..32767, always valid
    const int tx = tileId & 31;
    const int ty = (tileId >> 5) & 31;
    const int b  = tileId >> 10;
    const int x0 = tx * 16, y0 = ty * 16;
    const int base = b * (IMG_H * IMG_W);

    // 1D gaussian from the 2D window: g[i] = w2d[5][i] / sqrt(w2d[5][5])  (exact)
    const float invg5 = 1.0f / sqrtf(win[60]);

    // ---- A operand: banded weight matrix G[m][k] = g[k-m], 0<=k-m<=10 ----
    // A k order chosen here (same as B): lane half h, VGPR v, slot p: K = 16h + 2v + p
    v16h aA;
    #pragma unroll
    for (int v = 0; v < 8; ++v) {
        #pragma unroll
        for (int p = 0; p < 2; ++p) {
            int K  = 16 * h + 2 * v + p;
            int d  = K - m;
            int dc = d < 0 ? 0 : (d > 10 ? 10 : d);
            float wgt = win[55 + dc] * invg5;            // g[dc]
            aA[2 * v + p] = (_Float16)((d >= 0 && d <= 10) ? wgt : 0.0f);
        }
    }

    // ---- Load input slab and pack into f16 B operands (zero-padded SAME) ----
    // B k order chosen here (same as A): lane = N (column), element e: K = 16h + e  -> row y0-5+K
    v16h h1[2], h2[2];
    #pragma unroll
    for (int n = 0; n < 2; ++n) {
        int col  = x0 - 5 + 16 * n + m;
        bool cok = (col >= 0) && (col < IMG_W);
        int colc = col < 0 ? 0 : (col > IMG_W - 1 ? IMG_W - 1 : col);
        #pragma unroll
        for (int j = 0; j < 16; ++j) {
            int row  = y0 - 5 + 16 * h + j;
            bool ok  = cok && (row >= 0) && (row < IMG_H);
            int rowc = row < 0 ? 0 : (row > IMG_H - 1 ? IMG_H - 1 : row);
            int idx  = base + rowc * IMG_W + colc;       // always in-bounds
            float a  = ok ? img1[idx] : 0.0f;
            float c  = ok ? img2[idx] : 0.0f;
            h1[n][j] = (_Float16)a;
            h2[n][j] = (_Float16)c;
        }
    }

    // ---- 5 sources: img1, img2, img1^2, img2^2, img1*img2 ----
    v8f Hs[5];
    #pragma unroll
    for (int s = 0; s < 5; ++s) {
        // Vertical pass: V = G x rows  (2 N-chunks of columns)
        #pragma unroll
        for (int n = 0; n < 2; ++n) {
            v16h src = (s == 0) ? h1[n]
                     : (s == 1) ? h2[n]
                     : (s == 2) ? (v16h)(h1[n] * h1[n])
                     : (s == 3) ? (v16h)(h2[n] * h2[n])
                                : (v16h)(h1[n] * h2[n]);
            v8f z = {};
            v8f vac = __builtin_amdgcn_wmma_f32_16x16x32_f16(
                false, aA, false, src, (short)0, z, false, false);
            // C/D layout: VGPR i -> M = i + 8h (row y), N = m (col offset 16n+m)
            #pragma unroll
            for (int i = 0; i < 8; ++i)
                vlds[wave][(i + 8 * h) * LSTRIDE + 16 * n + m] = vac[i];
        }
        __syncthreads();

        // Horizontal pass on transposed V:  Ht = G x Vt
        v16h bh;
        #pragma unroll
        for (int e = 0; e < 16; ++e)
            bh[e] = (_Float16)vlds[wave][m * LSTRIDE + 16 * h + e];
        v8f z = {};
        Hs[s] = __builtin_amdgcn_wmma_f32_16x16x32_f16(
            false, aA, false, bh, (short)0, z, false, false);
        __syncthreads();
    }

    // ---- SSIM map (transposed tile; irrelevant for the global mean) ----
    const float C1 = 1e-4f;   // 0.01^2
    const float C2 = 9e-4f;   // 0.03^2
    float acc = 0.0f;
    #pragma unroll
    for (int i = 0; i < 8; ++i) {
        float mu1 = Hs[0][i], mu2 = Hs[1][i];
        float xx  = Hs[2][i], yy  = Hs[3][i], xy = Hs[4][i];
        float mu1s = mu1 * mu1, mu2s = mu2 * mu2, m12 = mu1 * mu2;
        float s1 = xx - mu1s, s2 = yy - mu2s, s12 = xy - m12;
        float num = (2.0f * m12 + C1) * (2.0f * s12 + C2);
        float den = (mu1s + mu2s + C1) * (s1 + s2 + C2);
        acc += num / den;
    }

    // wave32 reduce
    #pragma unroll
    for (int off = 16; off > 0; off >>= 1)
        acc += __shfl_xor(acc, off);
    if (lane == 0) redsh[wave] = acc;
    __syncthreads();
    if (tid == 0) {
        float ssum = 0.0f;
        #pragma unroll
        for (int wv = 0; wv < WPB; ++wv) ssum += redsh[wv];
        blockSums[blockIdx.x] = ssum;
    }
}

__global__ __launch_bounds__(256)
void ssim_reduce_kernel(const float* __restrict__ ws, float* __restrict__ out,
                        int n, float scale)
{
    __shared__ float sh[256];
    float a = 0.0f;
    for (int i = threadIdx.x; i < n; i += 256) a += ws[i];
    sh[threadIdx.x] = a;
    __syncthreads();
    for (int s = 128; s > 0; s >>= 1) {
        if (threadIdx.x < s) sh[threadIdx.x] += sh[threadIdx.x + s];
        __syncthreads();
    }
    if (threadIdx.x == 0) out[0] = sh[0] * scale;
}

extern "C" void kernel_launch(void* const* d_in, const int* in_sizes, int n_in,
                              void* d_out, int out_size, void* d_ws, size_t ws_size,
                              hipStream_t stream)
{
    (void)in_sizes; (void)n_in; (void)out_size; (void)ws_size;
    const float* img1 = (const float*)d_in[0];
    const float* img2 = (const float*)d_in[1];
    const float* win  = (const float*)d_in[2];
    float* out = (float*)d_out;
    float* ws  = (float*)d_ws;   // NBLOCKS floats of scratch

    ssim_wmma_kernel<<<NBLOCKS, 256, 0, stream>>>(img1, img2, win, ws);
    ssim_reduce_kernel<<<1, 256, 0, stream>>>(
        ws, out, NBLOCKS, 1.0f / (float)(BATCH * IMG_H * IMG_W));
}
